// DecoderRNN_10325101380224
// MI455X (gfx1250) — compile-verified, matrix operand order checked by reading, not run on hardware
//
#include <hip/hip_runtime.h>

// ---------------------------------------------------------------- constants
#define E_   512
#define H_   512
#define V_   50257
#define VPAD 50272          // V rounded up to 16
#define B_   32
#define T_   64
#define S_   2048           // T*B serial steps
#define G4H  2048           // 4*H gate rows
#define NWG  64             // persistent workgroups for the scan

typedef __attribute__((ext_vector_type(2)))  float   v2f;
typedef __attribute__((ext_vector_type(8)))  float   v8f;
typedef __attribute__((ext_vector_type(16))) __bf16  v16bf;

// =============================================================== kernel 1
// Build xs (S, E): xs[t*B+b] = (t==0) ? features[b] : embed_w[captions[b][t]]
__global__ void build_xs(const float* __restrict__ features,
                         const int*   __restrict__ captions,
                         const float* __restrict__ embed_w,
                         float* __restrict__ xs) {
    const int s = blockIdx.x;            // 0..S-1
    const int t = s >> 5;                // s / B
    const int b = s & 31;                // s % B
    const float* src = (t == 0)
        ? (features + (size_t)b * E_)
        : (embed_w + (size_t)captions[b * T_ + t] * E_);
    float* dst = xs + (size_t)s * E_;
    for (int e = threadIdx.x; e < E_; e += blockDim.x) dst[e] = src[e];
}

// =============================================================== kernel 2
// X0 = xs @ w_ih0^T + (b_ih0 + b_hh0)   -- f32 WMMA 16x16x4, M=N=2048, K=512
__global__ void x0_gemm(const float* __restrict__ xs,
                        const float* __restrict__ w_ih0,
                        const float* __restrict__ b_ih0,
                        const float* __restrict__ b_hh0,
                        float* __restrict__ X0) {
    const int wave = threadIdx.x >> 5;
    const int lane = threadIdx.x & 31;
    const int tile = blockIdx.x * 8 + wave;     // 128*128 = 16384 tiles
    const int mt = tile & 127;                  // seq tile
    const int nt = tile >> 7;                   // gate tile
    const int m  = lane & 15;
    const int g  = lane >> 4;

    const float* arow = xs    + (size_t)(mt * 16 + m) * E_;  // A[M][k]
    const float* brow = w_ih0 + (size_t)(nt * 16 + m) * E_;  // B[k][n] = w_ih0[n][k]
    v8f c = {};
    for (int k0 = 0; k0 < E_; k0 += 4) {
        v2f a, b;
        a.x = arow[k0 + g * 2 + 0]; a.y = arow[k0 + g * 2 + 1];
        b.x = brow[k0 + g * 2 + 0]; b.y = brow[k0 + g * 2 + 1];
        c = __builtin_amdgcn_wmma_f32_16x16x4_f32(false, a, false, b,
                                                  (short)0, c, false, false);
    }
    const int j = nt * 16 + m;                  // gate row (N)
    const float bias = b_ih0[j] + b_hh0[j];
#pragma unroll
    for (int r = 0; r < 8; ++r) {
        const int M = mt * 16 + r + g * 8;      // C: vgpr r, lanes0-15 M=r, 16-31 M=r+8
        X0[(size_t)M * G4H + j] = c[r] + bias;
    }
}

// =============================================================== kernel 3
// fc_w (V,H) f32 -> bf16, padded with zeros to VPAD rows
__global__ void cvt_fcw(const float* __restrict__ fcw, __bf16* __restrict__ out) {
    const size_t total = (size_t)VPAD * H_;
    const size_t valid = (size_t)V_ * H_;
    for (size_t i = (size_t)blockIdx.x * blockDim.x + threadIdx.x; i < total;
         i += (size_t)gridDim.x * blockDim.x)
        out[i] = (i < valid) ? (__bf16)fcw[i] : (__bf16)0.0f;
}

// =============================================================== kernel 4
// Zero the double-buffered h state and grid-barrier counters (every call).
__global__ void init_state(float* __restrict__ h0b, float* __restrict__ h1b,
                           unsigned* __restrict__ bar) {
    const int i = blockIdx.x * blockDim.x + threadIdx.x;
    if (i < 2 * H_) { h0b[i] = 0.0f; h1b[i] = 0.0f; }
    if (i < 8) bar[i] = 0u;
}

// ---------------------------------------------------------------- barrier
__device__ __forceinline__ void grid_barrier(unsigned* bar, unsigned nwg) {
    __syncthreads();
    if (threadIdx.x == 0) {
        volatile unsigned* vgen = bar + 1;
        const unsigned my = *vgen;
        __threadfence();
        const unsigned prev = atomicAdd(bar, 1u);
        if (prev == nwg - 1u) {
            bar[0] = 0u;                  // others only touch after gen bump
            __threadfence();
            atomicAdd(bar + 1, 1u);
        } else {
            while (*vgen == my) __builtin_amdgcn_s_sleep(1);
        }
        __threadfence();
    }
    __syncthreads();
}

// ---------------------------------------------------------------- LSTM cell
__device__ __forceinline__ float lstm_elem(float gi, float gf, float gg,
                                           float go, float& c) {
    const float i_ = 1.0f / (1.0f + expf(-gi));
    const float f_ = 1.0f / (1.0f + expf(-gf));
    const float o_ = 1.0f / (1.0f + expf(-go));
    c = f_ * c + i_ * tanhf(gg);
    return o_ * tanhf(c);
}

// =============================================================== kernel 5
// Persistent software-pipelined LSTM scan, ONE grid barrier per step.
// Iteration i (after barrier publishing h0(i) and h1(i-1)) computes BOTH
//   h0(i+1) = cell0(X0[i+1], h0(i))           [layer-0 recurrence]
//   h1(i)   = cell1(h0(i),  h1(i-1))          [layer-1, the output]
// then publishes both behind a single barrier.
// WG w owns h-elements [w*8, w*8+8); its 32 gate rows are {q*512 + w*8 + e}.
// Weights (12 MB) stay L2-resident; h vectors staged in LDS; c-state never
// leaves the owning WG's LDS.
__global__ void lstm_scan(const float* __restrict__ X0,
                          const float* __restrict__ w_hh0,
                          const float* __restrict__ w_ih1,
                          const float* __restrict__ w_hh1,
                          const float* __restrict__ b_ih1,
                          const float* __restrict__ b_hh1,
                          float* __restrict__ h0buf,    // [2][H]
                          float* __restrict__ h1buf,    // [2][H]
                          __bf16* __restrict__ htop,    // [S][H]
                          unsigned* __restrict__ bar) {
    __shared__ float lh0[H_];
    __shared__ float lh1[H_];
    __shared__ float sg0[32];
    __shared__ float sg1[32];
    __shared__ float c0l[8];
    __shared__ float c1l[8];

    const int tid = threadIdx.x;
    const int wg  = blockIdx.x;
    const int row = tid >> 3;              // 0..31 (local gate row)
    const int sub = tid & 7;               // 8 threads cooperate per row
    const int q   = row >> 3;              // gate index 0..3 (i,f,g,o)
    const int ml  = row & 7;               // local h element
    const int j   = q * H_ + wg * 8 + ml;  // global gate row

    // ---- prime: h0(0) = cell0(X0[0], h=0, c=0); h1(-1) = 0 (init kernel)
    if (tid < 8) {
        const int e  = wg * 8 + tid;
        const float gi = X0[0 * H_ + e];          // X0[0][0*512 + e]
        const float gf = X0[1 * H_ + e];          // X0[0][1*512 + e]
        const float gg = X0[2 * H_ + e];
        const float go = X0[3 * H_ + e];
        float c = 0.0f;
        h0buf[e] = lstm_elem(gi, gf, gg, go, c);  // parity 0
        c0l[tid] = c;
        c1l[tid] = 0.0f;
    }
    __threadfence();
    grid_barrier(bar, NWG);

    for (int i = 0; i < S_; ++i) {
        const float* h0c = h0buf + (i & 1) * H_;        // h0(i)   (published)
        float*       h0n = h0buf + ((i + 1) & 1) * H_;  // h0(i+1) (to write)
        const float* h1p = h1buf + ((i + 1) & 1) * H_;  // h1(i-1) (published)
        float*       h1c = h1buf + (i & 1) * H_;        // h1(i)   (to write)

        for (int k = tid; k < H_; k += 256) { lh0[k] = h0c[k]; lh1[k] = h1p[k]; }
        __syncthreads();

        // fused dots: layer-1 of step i, layer-0 of step i+1
        {
            const float* wi = w_ih1 + (size_t)j * H_;
            const float* wh = w_hh1 + (size_t)j * H_;
            const float* w0 = w_hh0 + (size_t)j * H_;
            float acc1 = 0.0f, acc0 = 0.0f;
#pragma unroll 8
            for (int u = 0; u < 64; ++u) {
                const int k = u * 8 + sub;
                const float h0k = lh0[k];
                acc1 += wi[k] * h0k + wh[k] * lh1[k];
                acc0 += w0[k] * h0k;
            }
            acc1 += __shfl_down(acc1, 4, 8);
            acc0 += __shfl_down(acc0, 4, 8);
            acc1 += __shfl_down(acc1, 2, 8);
            acc0 += __shfl_down(acc0, 2, 8);
            acc1 += __shfl_down(acc1, 1, 8);
            acc0 += __shfl_down(acc0, 1, 8);
            if (sub == 0) {
                sg1[row] = acc1 + b_ih1[j] + b_hh1[j];
                if (i + 1 < S_)
                    sg0[row] = acc0 + X0[(size_t)(i + 1) * G4H + j];
            }
        }
        __syncthreads();

        if (tid < 8) {
            const int e = wg * 8 + tid;
            // layer-1 elementwise -> h1(i): the output row
            {
                float c = c1l[tid];
                const float h1n = lstm_elem(sg1[0 * 8 + tid], sg1[1 * 8 + tid],
                                            sg1[2 * 8 + tid], sg1[3 * 8 + tid], c);
                c1l[tid] = c;
                h1c[e] = h1n;
                htop[(size_t)i * H_ + e] = (__bf16)h1n;
            }
            // layer-0 elementwise -> h0(i+1)
            if (i + 1 < S_) {
                float c = c0l[tid];
                const float h0v = lstm_elem(sg0[0 * 8 + tid], sg0[1 * 8 + tid],
                                            sg0[2 * 8 + tid], sg0[3 * 8 + tid], c);
                c0l[tid] = c;
                h0n[e] = h0v;
            }
        }
        __threadfence();
        grid_barrier(bar, NWG);     // single publish point per step
    }
}

// =============================================================== kernel 6
// logits = htop @ fc_w^T + fc_b, scattered to out[b][t][v].
// bf16 WMMA 16x16x32, f32 accumulate. M=2048, N=VPAD, K=512.
__global__ void fc_gemm(const __bf16* __restrict__ htop,
                        const __bf16* __restrict__ fcw,
                        const float*  __restrict__ fc_b,
                        float* __restrict__ out) {
    const int wave = threadIdx.x >> 5;
    const int lane = threadIdx.x & 31;
    const int tile = blockIdx.x * 8 + wave;     // (VPAD/16)*(S/16) tiles
    const int mt = tile & 127;                  // seq tile
    const int nt = tile >> 7;                   // vocab tile
    const int m  = lane & 15;
    const int g  = lane >> 4;

    const int v = nt * 16 + m;                  // vocab col for this lane
    const __bf16* arow = htop + (size_t)(mt * 16 + m) * H_;
    const __bf16* bcol = fcw  + (size_t)v * H_;          // B[k][n] = fcw[n][k]

    v8f c = {};
    for (int kt = 0; kt < H_; kt += 32) {
        v16bf a, b;
#pragma unroll
        for (int r = 0; r < 8; ++r) {
#pragma unroll
            for (int p = 0; p < 2; ++p) {
                const int k = ((r < 4) ? (r * 2 + p) : (16 + (r - 4) * 2 + p)) + g * 8;
                a[2 * r + p] = arow[kt + k];
                b[2 * r + p] = bcol[kt + k];
            }
        }
        c = __builtin_amdgcn_wmma_f32_16x16x32_bf16(false, a, false, b,
                                                    (short)0, c, false, false);
    }

    if (v < V_) {
        const float bias = fc_b[v];
#pragma unroll
        for (int r = 0; r < 8; ++r) {
            const int s  = mt * 16 + r + g * 8;  // serial step index
            const int t  = s >> 5;               // s / B
            const int bb = s & 31;               // s % B
            out[((size_t)bb * T_ + t) * V_ + v] = c[r] + bias;
        }
    }
}

// =============================================================== launcher
extern "C" void kernel_launch(void* const* d_in, const int* in_sizes, int n_in,
                              void* d_out, int out_size, void* d_ws, size_t ws_size,
                              hipStream_t stream) {
    (void)in_sizes; (void)n_in; (void)out_size; (void)ws_size;

    const float* features = (const float*)d_in[0];
    const int*   captions = (const int*)  d_in[1];
    const float* embed_w  = (const float*)d_in[2];
    const float* w_ih0    = (const float*)d_in[3];
    const float* w_hh0    = (const float*)d_in[4];
    const float* b_ih0    = (const float*)d_in[5];
    const float* b_hh0    = (const float*)d_in[6];
    const float* w_ih1    = (const float*)d_in[7];
    const float* w_hh1    = (const float*)d_in[8];
    const float* b_ih1    = (const float*)d_in[9];
    const float* b_hh1    = (const float*)d_in[10];
    const float* fc_w     = (const float*)d_in[11];
    const float* fc_b     = (const float*)d_in[12];
    float* out = (float*)d_out;

    // bump allocator over d_ws (256B aligned), total ~75 MB
    char*  ws  = (char*)d_ws;
    size_t off = 0;
    auto alloc = [&](size_t bytes) -> char* {
        char* p = ws + off;
        off += (bytes + 255) & ~(size_t)255;
        return p;
    };
    float*    xs    = (float*)   alloc((size_t)S_ * E_ * 4);        //  4 MB
    float*    X0    = (float*)   alloc((size_t)S_ * G4H * 4);       // 16 MB
    float*    h0b   = (float*)   alloc(2 * H_ * 4);
    float*    h1b   = (float*)   alloc(2 * H_ * 4);
    unsigned* bar   = (unsigned*)alloc(256);
    __bf16*   htop  = (__bf16*)  alloc((size_t)S_ * H_ * 2);        //  2 MB
    __bf16*   fcwb  = (__bf16*)  alloc((size_t)VPAD * H_ * 2);      // 51.5 MB

    build_xs  <<<S_,   256, 0, stream>>>(features, captions, embed_w, xs);
    x0_gemm   <<<2048, 256, 0, stream>>>(xs, w_ih0, b_ih0, b_hh0, X0);
    cvt_fcw   <<<2048, 256, 0, stream>>>(fc_w, fcwb);
    init_state<<<4,    256, 0, stream>>>(h0b, h1b, bar);
    lstm_scan <<<NWG,  256, 0, stream>>>(X0, w_hh0, w_ih1, w_hh1,
                                         b_ih1, b_hh1, h0b, h1b, htop, bar);
    // (VPAD/16 vocab tiles) * (S/16 seq tiles) / 8 waves = 50272 blocks
    fc_gemm   <<<(VPAD / 16) * (S_ / 16) / 8, 256, 0, stream>>>(htop, fcwb, fc_b, out);
}
